// MultiHeadAttention_73212012528234
// MI455X (gfx1250) — matrix operand order repaired and checked by reading, not compiled, not run on hardware
//
#include <hip/hip_runtime.h>
#include <hip/hip_bf16.h>

// Problem constants (B, S, D, H) = (8, 512, 1024, 16), DK = 64
constexpr int CB  = 8;
constexpr int CS  = 512;
constexpr int CD  = 1024;
constexpr int CH  = 16;
constexpr int CDK = 64;

typedef __attribute__((ext_vector_type(16))) __bf16 v16bf;
typedef __attribute__((ext_vector_type(8)))  __bf16 v8bf;
typedef __attribute__((ext_vector_type(8)))  float  v8f;

union BFrag { v16bf v; v8bf h[2]; };

// ---- Fragment loaders matching gfx1250 WMMA VGPR layouts (wave32) ----
// A matrix 16x32 bf16 (MxK): lanes 0-15 hold M=lane, K {0..7, 16..23};
// lanes 16-31 hold M=lane-16, K {8..15, 24..31}. Source row-major, K contiguous.
__device__ inline v16bf load_a_frag(const __bf16* base, int ld, int lane) {
  const int m  = lane & 15;
  const int kh = (lane >> 4) << 3;          // 0 or 8
  const __bf16* p = base + (size_t)m * ld + kh;
  BFrag f;
  f.h[0] = *(const v8bf*)(p);               // K kh .. kh+7     -> VGPR0..3
  f.h[1] = *(const v8bf*)(p + 16);          // K kh+16 .. kh+23 -> VGPR4..7
  return f.v;
}

// B matrix 32x16 bf16 (KxN): lanes 0-15 hold N=lane, K {0..7, 16..23};
// lanes 16-31 hold N=lane-16, K {8..15, 24..31}. Source rows = N, K contiguous (NT gemm).
__device__ inline v16bf load_b_frag(const __bf16* base, int ld, int lane) {
  const int n  = lane & 15;
  const int kh = (lane >> 4) << 3;          // 0 or 8
  const __bf16* p = base + (size_t)n * ld + kh;
  BFrag f;
  f.h[0] = *(const v8bf*)(p);
  f.h[1] = *(const v8bf*)(p + 16);
  return f.v;
}

// ---- fp32 -> bf16 staging (fold optional scale, e.g. 1/sqrt(DK) into W_Q) ----
__global__ __launch_bounds__(256) void cvt_f32_bf16(const float* __restrict__ src,
                                                    __bf16* __restrict__ dst,
                                                    int n, float scale) {
  int i = blockIdx.x * blockDim.x + threadIdx.x;
  if (i < n) dst[i] = (__bf16)(src[i] * scale);
}

// ---- bf16 NT GEMM:  out[b, m, n] = sum_k X[b, m, k] * W[n, k]
// Block = 128 threads (4 waves); each wave does a 16x64 slab (4 wmma tiles),
// workgroup covers 64x64 of the (S x D) output for one batch.
// mode 0: bf16 -> (B,H,S,DK)      (Q, K projections)
// mode 1: bf16 -> (B,H,DK,S)      (V, stored transposed for attn*V fragments)
// mode 2: f32  -> (B,S,D) d_out   (final projection)
__global__ __launch_bounds__(128) void gemm_bf16_nt(const __bf16* __restrict__ X,
                                                    const __bf16* __restrict__ W,
                                                    __bf16* __restrict__ obf,
                                                    float* __restrict__ of32,
                                                    int mode) {
  const int lane = threadIdx.x & 31;
  const int wave = threadIdx.x >> 5;
  const int b    = blockIdx.z;
  const int m0   = blockIdx.y * 64 + wave * 16;
  const int n0   = blockIdx.x * 64;

  const __bf16* Xb = X + ((size_t)b * CS + m0) * CD;
  const v8f zero = (v8f)0.0f;
  v8f acc[4];
#pragma unroll
  for (int t = 0; t < 4; ++t) acc[t] = zero;

  for (int k0 = 0; k0 < CD; k0 += 32) {
    __builtin_prefetch(Xb + k0 + 128, 0, 0);          // gfx1250 global_prefetch_b8
    v16bf a = load_a_frag(Xb + k0, CD, lane);
#pragma unroll
    for (int t = 0; t < 4; ++t) {
      v16bf bb = load_b_frag(W + (size_t)(n0 + t * 16) * CD + k0, CD, lane);
      acc[t] = __builtin_amdgcn_wmma_f32_16x16x32_bf16(false, a, false, bb,
                                                       (short)0, acc[t], false, false);
    }
  }

  // C/D layout: lanes 0-15 => N=lane, VGPR r = row M=r; lanes 16-31 => M=r+8
  const int hl = (lane >> 4) * 8;
  const int nc = lane & 15;
#pragma unroll
  for (int t = 0; t < 4; ++t) {
#pragma unroll
    for (int r = 0; r < 8; ++r) {
      const int m = m0 + r + hl;
      const int o = n0 + t * 16 + nc;
      const float v = acc[t][r];
      if (mode == 0) {
        const int h = o >> 6, dk = o & 63;
        obf[(((size_t)b * CH + h) * CS + m) * CDK + dk] = (__bf16)v;
      } else if (mode == 1) {
        const int h = o >> 6, dk = o & 63;
        obf[(((size_t)b * CH + h) * CDK + dk) * CS + m] = (__bf16)v;
      } else {
        of32[((size_t)b * CS + m) * CD + o] = v;
      }
    }
  }
}

// ---- Fused attention: one workgroup per (b, h, 16-row q tile) ----
// scores (WMMA) -> softmax -> conv-weighted recombine -> softmax -> attn out
// (f32 to d_out + bf16 to LDS) -> context = attn * V (WMMA, A from LDS).
#define SC_LD 520   // padded row stride to spread LDS banks

__global__ __launch_bounds__(128) void attn_kernel(const __bf16* __restrict__ Qb,
                                                   const __bf16* __restrict__ Kb,
                                                   const __bf16* __restrict__ Vt,
                                                   const unsigned char* __restrict__ mask,
                                                   const float* __restrict__ adj,
                                                   const float* __restrict__ dist,
                                                   const float* __restrict__ cw,
                                                   const float* __restrict__ cb,
                                                   float* __restrict__ attn_out,
                                                   __bf16* __restrict__ ctx) {
  __shared__ float  sc[16 * SC_LD];     // 16 x 512 scores (f32)
  __shared__ __bf16 ab[16 * CS];        // 16 x 512 attn (bf16) for context WMMA
  __shared__ float  red[16 * 8];        // row reductions

  const int lane = threadIdx.x & 31;
  const int wave = threadIdx.x >> 5;
  const int q0 = blockIdx.x * 16;
  const int h  = blockIdx.y;
  const int b  = blockIdx.z;
  const size_t bh = (size_t)b * CH + h;

  // ---- scores = (Q/sqrt(DK)) K^T : each wave -> 8 col tiles (128 cols) ----
  const __bf16* Qp = Qb + (bh * CS + q0) * CDK;
  const __bf16* Kp = Kb + bh * CS * CDK;
  const v16bf aq0 = load_a_frag(Qp,      CDK, lane);
  const v16bf aq1 = load_a_frag(Qp + 32, CDK, lane);

  const v8f zero = (v8f)0.0f;
  v8f acc[8];
#pragma unroll
  for (int t = 0; t < 8; ++t) acc[t] = zero;

  const int s_base = wave * 128;
#pragma unroll
  for (int t = 0; t < 8; ++t) {
    const __bf16* kb = Kp + (size_t)(s_base + t * 16) * CDK;
    v16bf b0 = load_b_frag(kb,      CDK, lane);
    acc[t] = __builtin_amdgcn_wmma_f32_16x16x32_bf16(false, aq0, false, b0,
                                                     (short)0, acc[t], false, false);
    v16bf b1 = load_b_frag(kb + 32, CDK, lane);
    acc[t] = __builtin_amdgcn_wmma_f32_16x16x32_bf16(false, aq1, false, b1,
                                                     (short)0, acc[t], false, false);
  }
  const int hl = (lane >> 4) * 8;
  const int nc = lane & 15;
#pragma unroll
  for (int t = 0; t < 8; ++t)
#pragma unroll
    for (int r = 0; r < 8; ++r)
      sc[(r + hl) * SC_LD + s_base + t * 16 + nc] = acc[t][r];
  __syncthreads();

  // ---- softmax pipeline: thread -> (row = tid>>3, 64-col segment = tid&7) ----
  const int row = threadIdx.x >> 3;
  const int seg = threadIdx.x & 7;
  float* sr = sc + row * SC_LD + seg * 64;
  const size_t mrow = ((size_t)b * CS + q0 + row) * CS + seg * 64;

  unsigned long long mb = 0;            // pack 64 mask bits
  for (int j = 0; j < 64; ++j)
    if (mask[mrow + j]) mb |= 1ull << j;

  // softmax #1 (masked scores)
  float mx = -3.4e38f;
  for (int j = 0; j < 64; ++j) {
    float s = ((mb >> j) & 1) ? -1.0e9f : sr[j];
    sr[j] = s;
    mx = fmaxf(mx, s);
  }
  red[row * 8 + seg] = mx;  __syncthreads();
  float rmax = red[row * 8];
  for (int j = 1; j < 8; ++j) rmax = fmaxf(rmax, red[row * 8 + j]);
  __syncthreads();
  float sum = 0.f;
  for (int j = 0; j < 64; ++j) { float e = __expf(sr[j] - rmax); sr[j] = e; sum += e; }
  red[row * 8 + seg] = sum; __syncthreads();
  float rsum = 0.f;
  for (int j = 0; j < 8; ++j) rsum += red[row * 8 + j];
  const float inv1 = 1.0f / rsum;
  __syncthreads();

  // weighted = cw0*softmax + cw1*exp(-dist)[!mask] + cw2*adj + cb; mask -> -1e9
  const float cw0 = cw[0], cw1 = cw[1], cw2 = cw[2], cb0 = cb[0];
  float mx2 = -3.4e38f;
  for (int j = 0; j < 64; ++j) {
    float w;
    if ((mb >> j) & 1) {
      w = -1.0e9f;
    } else {
      const float p = sr[j] * inv1;
      w = cw0 * p + cw1 * __expf(-dist[mrow + j]) + cw2 * adj[mrow + j] + cb0;
    }
    sr[j] = w;
    mx2 = fmaxf(mx2, w);
  }
  red[row * 8 + seg] = mx2; __syncthreads();
  float rmax2 = red[row * 8];
  for (int j = 1; j < 8; ++j) rmax2 = fmaxf(rmax2, red[row * 8 + j]);
  __syncthreads();
  float sum2 = 0.f;
  for (int j = 0; j < 64; ++j) { float e = __expf(sr[j] - rmax2); sr[j] = e; sum2 += e; }
  red[row * 8 + seg] = sum2; __syncthreads();
  float rsum2 = 0.f;
  for (int j = 0; j < 8; ++j) rsum2 += red[row * 8 + j];
  const float inv2 = 1.0f / rsum2;

  // attn: f32 -> d_out (streaming, written once), bf16 -> LDS for context WMMA
  float* aout = attn_out + (bh * CS + q0 + row) * CS + seg * 64;
  __bf16* abr = ab + row * CS + seg * 64;
  for (int j = 0; j < 64; ++j) {
    const float av = sr[j] * inv2;
    aout[j] = av;
    abr[j]  = (__bf16)av;
  }
  __syncthreads();

  // ---- context = attn * V : wave -> 16x16 tile (dk range wave*16..+16) ----
  const int n0 = wave * 16;
  v8f cacc = zero;
  const __bf16* Vp = Vt + (bh * CDK + n0) * CS;   // (B,H,DK,S): rows=dk, K=s contiguous
  for (int ks = 0; ks < CS; ks += 32) {
    v16bf aa = load_a_frag(ab + ks, CS, lane);    // from LDS (ds_load)
    v16bf bb = load_b_frag(Vp + ks, CS, lane);
    cacc = __builtin_amdgcn_wmma_f32_16x16x32_bf16(false, aa, false, bb,
                                                   (short)0, cacc, false, false);
  }
#pragma unroll
  for (int r = 0; r < 8; ++r)
    ctx[((size_t)b * CS + q0 + r + hl) * CD + h * CDK + n0 + nc] = (__bf16)cacc[r];
}

// ---------------------------------------------------------------------------
extern "C" void kernel_launch(void* const* d_in, const int* in_sizes, int n_in,
                              void* d_out, int out_size, void* d_ws, size_t ws_size,
                              hipStream_t stream) {
  (void)in_sizes; (void)n_in; (void)out_size; (void)ws_size;

  const float* in_Q  = (const float*)d_in[0];
  const float* in_K  = (const float*)d_in[1];
  const float* in_V  = (const float*)d_in[2];
  const unsigned char* mask = (const unsigned char*)d_in[3];  // jnp.bool_ = 1 byte
  const float* adj   = (const float*)d_in[4];
  const float* dist  = (const float*)d_in[5];
  const float* W_Q   = (const float*)d_in[6];
  const float* W_K   = (const float*)d_in[7];
  const float* W_V   = (const float*)d_in[8];
  const float* W_fc  = (const float*)d_in[9];
  const float* cw    = (const float*)d_in[10];
  const float* cb    = (const float*)d_in[11];

  const size_t nBSD = (size_t)CB * CS * CD;   // 4,194,304
  const size_t nDD  = (size_t)CD * CD;        // 1,048,576

  // workspace layout (bytes), all 256-aligned; total = 64 MB
  char* ws = (char*)d_ws;
  __bf16* bfQ  = (__bf16*)(ws);
  __bf16* bfK  = (__bf16*)(ws + 1 * nBSD * 2);
  __bf16* bfV  = (__bf16*)(ws + 2 * nBSD * 2);
  __bf16* wqb  = (__bf16*)(ws + 3 * nBSD * 2);
  __bf16* wkb  = (__bf16*)(ws + 3 * nBSD * 2 + 1 * nDD * 2);
  __bf16* wvb  = (__bf16*)(ws + 3 * nBSD * 2 + 2 * nDD * 2);
  __bf16* wfcb = (__bf16*)(ws + 3 * nBSD * 2 + 3 * nDD * 2);
  __bf16* Qbf  = (__bf16*)(ws + 3 * nBSD * 2 + 4 * nDD * 2);
  __bf16* Kbf  = (__bf16*)(ws + 4 * nBSD * 2 + 4 * nDD * 2);
  __bf16* Vtb  = (__bf16*)(ws + 5 * nBSD * 2 + 4 * nDD * 2);
  __bf16* ctx  = (__bf16*)(ws + 6 * nBSD * 2 + 4 * nDD * 2);

  // 1) fp32 -> bf16 staging (scale 1/sqrt(DK)=0.125 folded into W_Q)
  const int tb = 256;
  cvt_f32_bf16<<<dim3((nBSD + tb - 1) / tb), tb, 0, stream>>>(in_Q, bfQ, (int)nBSD, 1.0f);
  cvt_f32_bf16<<<dim3((nBSD + tb - 1) / tb), tb, 0, stream>>>(in_K, bfK, (int)nBSD, 1.0f);
  cvt_f32_bf16<<<dim3((nBSD + tb - 1) / tb), tb, 0, stream>>>(in_V, bfV, (int)nBSD, 1.0f);
  cvt_f32_bf16<<<dim3((nDD  + tb - 1) / tb), tb, 0, stream>>>(W_Q,  wqb, (int)nDD, 0.125f);
  cvt_f32_bf16<<<dim3((nDD  + tb - 1) / tb), tb, 0, stream>>>(W_K,  wkb, (int)nDD, 1.0f);
  cvt_f32_bf16<<<dim3((nDD  + tb - 1) / tb), tb, 0, stream>>>(W_V,  wvb, (int)nDD, 1.0f);
  cvt_f32_bf16<<<dim3((nDD  + tb - 1) / tb), tb, 0, stream>>>(W_fc, wfcb,(int)nDD, 1.0f);

  // 2) QKV projections (WMMA bf16, f32 accumulate)
  dim3 gg(CD / 64, CS / 64, CB);   // (16, 8, 8)
  gemm_bf16_nt<<<gg, 128, 0, stream>>>(bfQ, wqb, Qbf, nullptr, 0);
  gemm_bf16_nt<<<gg, 128, 0, stream>>>(bfK, wkb, Kbf, nullptr, 0);
  gemm_bf16_nt<<<gg, 128, 0, stream>>>(bfV, wvb, Vtb, nullptr, 1);  // V transposed

  // 3) fused attention (scores -> 2x softmax + recombine -> attn out + context)
  float* out_f32  = (float*)d_out;                 // (B,S,D) output
  float* attn_out = out_f32 + nBSD;                // (B,H,S,S) attn
  dim3 ga(CS / 16, CH, CB);        // (32, 16, 8)
  attn_kernel<<<ga, 128, 0, stream>>>(Qbf, Kbf, Vtb, mask, adj, dist, cw, cb,
                                      attn_out, ctx);

  // 4) output projection -> d_out (f32)
  gemm_bf16_nt<<<gg, 128, 0, stream>>>(ctx, wfcb, nullptr, out_f32, 2);
}
